// CausalSelfAttention_48893907698345
// MI455X (gfx1250) — matrix operand order repaired and checked by reading, compiled, not run on hardware
//
#include <hip/hip_runtime.h>
#include <hip/hip_bf16.h>

// Problem constants (from reference)
constexpr int Bb = 2;
constexpr int Tt = 2048;
constexpr int Dd = 512;
constexpr int Hh = 8;

// Tiling
constexpr int QROWS = 32;    // query rows per workgroup
constexpr int KBLK  = 64;    // key block per flash iteration

// LDS strides (elements); multiples of 8 -> 16B-aligned b128 fragment reads,
// odd-ish dword phases -> distinct banks across 16-lane fragment reads.
constexpr int QS  = 520;     // Q tile:   [32 rows][512 + pad] (bf16)
constexpr int SS  = 65;      // S tile:   [32 rows][64 + pad]  (f32)
constexpr int PS  = 72;      // P tile:   [32 rows][64 + pad]  (bf16)

// fallback-path strides
constexpr int ATS = 40;      // A^T tile: [512 cols][32 k + pad]
constexpr int VTS = 72;      // V^T tile: [512 d][64 keys + pad]

typedef __attribute__((ext_vector_type(16))) __bf16 v16bf;
typedef __attribute__((ext_vector_type(8)))  float  v8f;

union Frag {
  v16bf v;
  unsigned short us[16];
  uint4 q4[2];
};

__device__ __forceinline__ unsigned short f2bf(float f) {
  union { float f; unsigned u; } x; x.f = f;
  unsigned u = x.u;
  unsigned r = u + 0x7FFFu + ((u >> 16) & 1u);   // round-to-nearest-even
  return (unsigned short)(r >> 16);
}

__device__ __forceinline__ void cvt4(unsigned short* d, float4 f) {
  d[0] = f2bf(f.x); d[1] = f2bf(f.y); d[2] = f2bf(f.z); d[3] = f2bf(f.w);
}

__device__ __forceinline__ v8f wmma_bf16(const Frag& a, const Frag& b, v8f c) {
  return __builtin_amdgcn_wmma_f32_16x16x32_bf16(
      false, a.v, false, b.v, (short)0, c, false, false);
}

// ---------------------------------------------------------------------------
// Prep: tiled f32 -> bf16 convert + transpose.
//   src   : [batch][R][C] f32
//   dstT  : [batch][C][R] bf16 bits
//   dstC  : [batch][R][C] bf16 bits (optional straight copy, may be null)
// grid = (R/64, C/64, batch), block = 256
// ---------------------------------------------------------------------------
__global__ __launch_bounds__(256)
void cvt_transpose(const float* __restrict__ src,
                   unsigned short* __restrict__ dstT,
                   unsigned short* __restrict__ dstC,
                   int R, int C) {
  __shared__ unsigned short tile[64][68];
  const int tid = threadIdx.x;
  const int i0  = tid >> 4;          // 0..15
  const int j   = (tid & 15) << 2;   // 0..60 step 4
  const size_t bofs = (size_t)blockIdx.z * R * C;
  const float* s = src + bofs;
  unsigned short* dT = dstT + bofs;
  const int r0 = blockIdx.x * 64;
  const int c0 = blockIdx.y * 64;

  #pragma unroll
  for (int rr = 0; rr < 4; ++rr) {
    int i = i0 + rr * 16;
    float4 f = *reinterpret_cast<const float4*>(s + (size_t)(r0 + i) * C + c0 + j);
    unsigned short b0 = f2bf(f.x), b1 = f2bf(f.y), b2 = f2bf(f.z), b3 = f2bf(f.w);
    tile[i][j + 0] = b0; tile[i][j + 1] = b1;
    tile[i][j + 2] = b2; tile[i][j + 3] = b3;
    if (dstC) {
      union { uint2 u2; unsigned short s4[4]; } o;
      o.s4[0] = b0; o.s4[1] = b1; o.s4[2] = b2; o.s4[3] = b3;
      *reinterpret_cast<uint2*>(dstC + bofs + (size_t)(r0 + i) * C + c0 + j) = o.u2;
    }
  }
  __syncthreads();
  #pragma unroll
  for (int cc = 0; cc < 4; ++cc) {
    int c = i0 + cc * 16;            // column of src tile = row of dstT
    union { uint2 u2; unsigned short s4[4]; } o;
    o.s4[0] = tile[j + 0][c]; o.s4[1] = tile[j + 1][c];
    o.s4[2] = tile[j + 2][c]; o.s4[3] = tile[j + 3][c];
    *reinterpret_cast<uint2*>(dT + (size_t)(c0 + c) * R + r0 + j) = o.u2;
  }
}

// ---------------------------------------------------------------------------
// Fast path: fused bilinear flash attention, all operands pre-baked bf16.
//   Hbf  [b][t][d], HbfT [b][d][t], AbfT [head][e][d]
// grid = (T/32, H, B), block = 256 (8 wave32)
// ---------------------------------------------------------------------------
struct SMemFast {
  unsigned short Q[QROWS * QS];    // Q tile (bf16 bits)
  float          S[QROWS * SS];    // raw scores
  unsigned short P[QROWS * PS];    // exp'd probs (bf16 bits)
  float          pmax[QROWS * 8];  // partial row maxima (8 chunks/row)
  float          psum[QROWS * 8];  // partial row sums
  float          m[QROWS];
  float          l[QROWS];
  float          c[QROWS];
};

__global__ __launch_bounds__(256)
void attn_fast(const unsigned short* __restrict__ Hbf,
               const unsigned short* __restrict__ HbfT,
               const unsigned short* __restrict__ AbfT,
               float* __restrict__ out) {
  __shared__ SMemFast sh;

  const int tid   = threadIdx.x;
  const int wave  = tid >> 5;
  const int lane  = tid & 31;
  const int l16   = lane & 15;
  const int lhalf = lane >> 4;
  const int wm = wave & 1;            // 16-row half of the 32-row tile
  const int wn = wave >> 1;           // 0..3 column group

  // softmax work decomposition: each thread owns (row, 8-col chunk)
  const int srow = tid & 31;
  const int scc  = tid >> 5;          // 0..7

  const int qb   = blockIdx.x;
  const int head = blockIdx.y;
  const int b    = blockIdx.z;
  const int q0   = qb * QROWS;

  const v8f vzero = {0.f,0.f,0.f,0.f,0.f,0.f,0.f,0.f};
  if (tid < QROWS) { sh.m[tid] = -3.0e38f; sh.l[tid] = 0.f; }

  // ---------- Phase A: Q tile = h_block @ A_head, all-global operands ------
  v8f accQ[8];
  #pragma unroll
  for (int i = 0; i < 8; ++i) accQ[i] = vzero;

  const unsigned short* hrow =
      Hbf + (size_t)(b * Tt + q0 + wm * 16 + l16) * Dd;

  for (int kk = 0; kk < Dd / 32; ++kk) {
    const int dbase = kk * 32;
    Frag qa;
    const unsigned short* hp = hrow + dbase + lhalf * 8;
    qa.q4[0] = *reinterpret_cast<const uint4*>(hp);
    qa.q4[1] = *reinterpret_cast<const uint4*>(hp + 16);
    #pragma unroll
    for (int nt = 0; nt < 8; ++nt) {
      Frag bf;
      const unsigned short* bp =
          AbfT + (size_t)(head * Dd + wn * 128 + nt * 16 + l16) * Dd
               + dbase + lhalf * 8;
      bf.q4[0] = *reinterpret_cast<const uint4*>(bp);
      bf.q4[1] = *reinterpret_cast<const uint4*>(bp + 16);
      accQ[nt] = wmma_bf16(qa, bf, accQ[nt]);
    }
  }

  // store Q tile to LDS (bf16). C layout: row = g + 8*lhalf (+wm*16), col=l16
  #pragma unroll
  for (int nt = 0; nt < 8; ++nt) {
    #pragma unroll
    for (int g = 0; g < 8; ++g) {
      int row = wm * 16 + g + 8 * lhalf;
      int col = wn * 128 + nt * 16 + l16;
      sh.Q[row * QS + col] = f2bf(accQ[nt][g]);
    }
  }
  __syncthreads();

  // ---------- Phase B: flash attention over key blocks ---------------------
  v8f accO[8];
  #pragma unroll
  for (int i = 0; i < 8; ++i) accO[i] = vzero;

  const int nkb = (q0 + QROWS + KBLK - 1) / KBLK;

  for (int kb = 0; kb < nkb; ++kb) {
    const int k0 = kb * KBLK;

    // prefetch next key block (global_prefetch_b8)
    if (kb + 1 < nkb) {
      __builtin_prefetch(Hbf + (size_t)(b * Tt + k0 + KBLK + lane) * Dd, 0, 1);
      __builtin_prefetch(HbfT + (size_t)(b * Dd + wn * 128 + lane) * Tt + k0 + KBLK, 0, 1);
    }

    // S = Q K^T; wave (wm, wn) computes 16x16 tile
    v8f accS = vzero;
    const unsigned short* krow =
        Hbf + (size_t)(b * Tt + k0 + wn * 16 + l16) * Dd;
    for (int kk = 0; kk < Dd / 32; ++kk) {
      const int dbase = kk * 32;
      Frag qa;
      const unsigned short* qp =
          &sh.Q[(wm * 16 + l16) * QS + dbase + lhalf * 8];
      qa.q4[0] = *reinterpret_cast<const uint4*>(qp);
      qa.q4[1] = *reinterpret_cast<const uint4*>(qp + 16);
      Frag kf;
      const unsigned short* kp = krow + dbase + lhalf * 8;
      kf.q4[0] = *reinterpret_cast<const uint4*>(kp);
      kf.q4[1] = *reinterpret_cast<const uint4*>(kp + 16);
      accS = wmma_bf16(qa, kf, accS);
    }

    #pragma unroll
    for (int g = 0; g < 8; ++g)
      sh.S[(wm * 16 + g + 8 * lhalf) * SS + wn * 16 + l16] = accS[g];
    __syncthreads();

    // ---- parallel online softmax over all 256 threads ----
    // pass 1: per-(row, 8-col chunk) partial max
    {
      const int q = q0 + srow;
      float pm = -3.0e38f;
      #pragma unroll
      for (int c8 = 0; c8 < 8; ++c8) {
        int cidx = scc * 8 + c8;
        float v = sh.S[srow * SS + cidx];
        if (k0 + cidx > q) v = -3.0e38f;
        pm = fmaxf(pm, v);
      }
      sh.pmax[srow * 8 + scc] = pm;
    }
    __syncthreads();
    if (tid < QROWS) {
      const int r = tid;
      float mOld = sh.m[r];
      float mNew = mOld;
      #pragma unroll
      for (int cc = 0; cc < 8; ++cc) mNew = fmaxf(mNew, sh.pmax[r * 8 + cc]);
      float corr = __expf(mOld - mNew);
      sh.c[r] = corr;
      sh.m[r] = mNew;
      sh.l[r] = sh.l[r] * corr;      // partial; chunk sums added below
    }
    __syncthreads();
    // pass 2: exponentiate, store P (bf16), partial sums
    {
      const int q = q0 + srow;
      float mNew = sh.m[srow];
      float ps = 0.f;
      #pragma unroll
      for (int c8 = 0; c8 < 8; ++c8) {
        int cidx = scc * 8 + c8;
        float v = sh.S[srow * SS + cidx];
        if (k0 + cidx > q) v = -3.0e38f;
        float p = __expf(v - mNew);
        ps += p;
        sh.P[srow * PS + cidx] = f2bf(p);
      }
      sh.psum[srow * 8 + scc] = ps;
    }
    __syncthreads();
    if (tid < QROWS) {
      const int r = tid;
      float ls = sh.l[r];
      #pragma unroll
      for (int cc = 0; cc < 8; ++cc) ls += sh.psum[r * 8 + cc];
      sh.l[r] = ls;                  // visible at epilogue (barrier after loop)
    }

    // rescale O, then O += P @ V  (V fragments straight from HbfT global)
    #pragma unroll
    for (int g = 0; g < 8; ++g) {
      float corr = sh.c[wm * 16 + g + 8 * lhalf];
      #pragma unroll
      for (int nt = 0; nt < 8; ++nt) accO[nt][g] *= corr;
    }

    #pragma unroll
    for (int kc = 0; kc < 2; ++kc) {
      Frag pa;
      const unsigned short* pp =
          &sh.P[(wm * 16 + l16) * PS + kc * 32 + lhalf * 8];
      pa.q4[0] = *reinterpret_cast<const uint4*>(pp);
      pa.q4[1] = *reinterpret_cast<const uint4*>(pp + 16);
      #pragma unroll
      for (int nt = 0; nt < 8; ++nt) {
        Frag vf;
        const unsigned short* vp =
            HbfT + (size_t)(b * Dd + wn * 128 + nt * 16 + l16) * Tt
                 + k0 + kc * 32 + lhalf * 8;
        vf.q4[0] = *reinterpret_cast<const uint4*>(vp);
        vf.q4[1] = *reinterpret_cast<const uint4*>(vp + 16);
        accO[nt] = wmma_bf16(pa, vf, accO[nt]);
      }
    }
  }

  // ---------- epilogue -----------------------------------------------------
  __syncthreads();                   // make final sh.l visible to all waves
  #pragma unroll
  for (int g = 0; g < 8; ++g) {
    int row  = wm * 16 + g + 8 * lhalf;
    float li = 1.0f / sh.l[row];
    int t    = q0 + row;
    #pragma unroll
    for (int nt = 0; nt < 8; ++nt) {
      int col = head * Dd + wn * 128 + nt * 16 + l16;
      out[(size_t)(b * Tt + t) * (Hh * Dd) + col] = accO[nt][g] * li;
    }
  }
}

// ---------------------------------------------------------------------------
// Fallback path (no workspace): fused kernel, converts on the fly.
// ---------------------------------------------------------------------------
struct SMemFb {
  union {
    unsigned short AT[Dd * ATS];
    unsigned short VT[Dd * VTS];
  };
  unsigned short Q[QROWS * QS];
  float          S[QROWS * SS];
  unsigned short P[QROWS * PS];
  float          m[QROWS];
  float          l[QROWS];
  float          c[QROWS];
};

__global__ __launch_bounds__(256)
void attn_fallback(const float* __restrict__ h,
                   const float* __restrict__ A,
                   float* __restrict__ out) {
  __shared__ SMemFb sh;
  const int tid   = threadIdx.x;
  const int wave  = tid >> 5;
  const int lane  = tid & 31;
  const int l16   = lane & 15;
  const int lhalf = lane >> 4;
  const int wm = wave & 1;
  const int wn = wave >> 1;
  const int qb   = blockIdx.x;
  const int head = blockIdx.y;
  const int b    = blockIdx.z;
  const int q0   = qb * QROWS;
  const v8f vzero = {0.f,0.f,0.f,0.f,0.f,0.f,0.f,0.f};

  if (tid < QROWS) { sh.m[tid] = -3.0e38f; sh.l[tid] = 0.f; }

  v8f accQ[8];
  #pragma unroll
  for (int i = 0; i < 8; ++i) accQ[i] = vzero;
  const float* Ah = A + (size_t)head * Dd * Dd;

  for (int kk = 0; kk < Dd / 32; ++kk) {
    const int dbase = kk * 32;
    __syncthreads();
    const float4* Asrc = reinterpret_cast<const float4*>(Ah + (size_t)dbase * Dd);
    #pragma unroll
    for (int j = 0; j < 16; ++j) {
      int idx4 = tid + j * 256;
      int k = idx4 >> 7;
      int n = (idx4 & 127) << 2;
      float4 f = Asrc[idx4];
      sh.AT[(n + 0) * ATS + k] = f2bf(f.x);
      sh.AT[(n + 1) * ATS + k] = f2bf(f.y);
      sh.AT[(n + 2) * ATS + k] = f2bf(f.z);
      sh.AT[(n + 3) * ATS + k] = f2bf(f.w);
    }
    __syncthreads();
    Frag qa;
    const float* hp = h + (size_t)(b * Tt + q0 + wm * 16 + l16) * Dd
                        + dbase + lhalf * 8;
    cvt4(&qa.us[0],  *reinterpret_cast<const float4*>(hp));
    cvt4(&qa.us[4],  *reinterpret_cast<const float4*>(hp + 4));
    cvt4(&qa.us[8],  *reinterpret_cast<const float4*>(hp + 16));
    cvt4(&qa.us[12], *reinterpret_cast<const float4*>(hp + 20));
    #pragma unroll
    for (int nt = 0; nt < 8; ++nt) {
      Frag bf;
      const unsigned short* bp =
          &sh.AT[(wn * 128 + nt * 16 + l16) * ATS + lhalf * 8];
      bf.q4[0] = *reinterpret_cast<const uint4*>(bp);
      bf.q4[1] = *reinterpret_cast<const uint4*>(bp + 16);
      accQ[nt] = wmma_bf16(qa, bf, accQ[nt]);
    }
  }

  __syncthreads();
  #pragma unroll
  for (int nt = 0; nt < 8; ++nt)
    #pragma unroll
    for (int g = 0; g < 8; ++g)
      sh.Q[(wm * 16 + g + 8 * lhalf) * QS + wn * 128 + nt * 16 + l16] =
          f2bf(accQ[nt][g]);
  __syncthreads();

  v8f accO[8];
  #pragma unroll
  for (int i = 0; i < 8; ++i) accO[i] = vzero;
  const int nkb = (q0 + QROWS + KBLK - 1) / KBLK;

  for (int kb = 0; kb < nkb; ++kb) {
    const int k0 = kb * KBLK;
    __syncthreads();
    const float4* Hsrc =
        reinterpret_cast<const float4*>(h + (size_t)(b * Tt + k0) * Dd);
    #pragma unroll
    for (int j = 0; j < 32; ++j) {
      int idx4 = tid + j * 256;
      int k = idx4 >> 7;
      int d = (idx4 & 127) << 2;
      float4 f = Hsrc[idx4];
      sh.VT[(d + 0) * VTS + k] = f2bf(f.x);
      sh.VT[(d + 1) * VTS + k] = f2bf(f.y);
      sh.VT[(d + 2) * VTS + k] = f2bf(f.z);
      sh.VT[(d + 3) * VTS + k] = f2bf(f.w);
    }
    v8f accS = vzero;
    const int key = k0 + wn * 16 + l16;
    #pragma unroll
    for (int kk = 0; kk < Dd / 32; ++kk) {
      const int dbase = kk * 32;
      Frag qa;
      const unsigned short* qp =
          &sh.Q[(wm * 16 + l16) * QS + dbase + lhalf * 8];
      qa.q4[0] = *reinterpret_cast<const uint4*>(qp);
      qa.q4[1] = *reinterpret_cast<const uint4*>(qp + 16);
      Frag kf;
      const float* kp = h + (size_t)(b * Tt + key) * Dd + dbase + lhalf * 8;
      cvt4(&kf.us[0],  *reinterpret_cast<const float4*>(kp));
      cvt4(&kf.us[4],  *reinterpret_cast<const float4*>(kp + 4));
      cvt4(&kf.us[8],  *reinterpret_cast<const float4*>(kp + 16));
      cvt4(&kf.us[12], *reinterpret_cast<const float4*>(kp + 20));
      accS = wmma_bf16(qa, kf, accS);
    }
    #pragma unroll
    for (int g = 0; g < 8; ++g)
      sh.S[(wm * 16 + g + 8 * lhalf) * SS + wn * 16 + l16] = accS[g];
    __syncthreads();
    if (tid < QROWS) {
      const int r = tid;
      const int q = q0 + r;
      float mOld = sh.m[r];
      float mNew = mOld;
      for (int cidx = 0; cidx < KBLK; ++cidx) {
        float v = sh.S[r * SS + cidx];
        if (k0 + cidx > q) v = -3.0e38f;
        mNew = fmaxf(mNew, v);
      }
      float corr = __expf(mOld - mNew);
      float lsum = sh.l[r] * corr;
      for (int cidx = 0; cidx < KBLK; ++cidx) {
        float v = sh.S[r * SS + cidx];
        if (k0 + cidx > q) v = -3.0e38f;
        float p = __expf(v - mNew);
        lsum += p;
        sh.P[r * PS + cidx] = f2bf(p);
      }
      sh.m[r] = mNew; sh.l[r] = lsum; sh.c[r] = corr;
    }
    __syncthreads();
    #pragma unroll
    for (int g = 0; g < 8; ++g) {
      float corr = sh.c[wm * 16 + g + 8 * lhalf];
      #pragma unroll
      for (int nt = 0; nt < 8; ++nt) accO[nt][g] *= corr;
    }
    #pragma unroll
    for (int kc = 0; kc < 2; ++kc) {
      Frag pa;
      const unsigned short* pp =
          &sh.P[(wm * 16 + l16) * PS + kc * 32 + lhalf * 8];
      pa.q4[0] = *reinterpret_cast<const uint4*>(pp);
      pa.q4[1] = *reinterpret_cast<const uint4*>(pp + 16);
      #pragma unroll
      for (int nt = 0; nt < 8; ++nt) {
        Frag vf;
        const unsigned short* vp =
            &sh.VT[(wn * 128 + nt * 16 + l16) * VTS + kc * 32 + lhalf * 8];
        vf.q4[0] = *reinterpret_cast<const uint4*>(vp);
        vf.q4[1] = *reinterpret_cast<const uint4*>(vp + 16);
        accO[nt] = wmma_bf16(pa, vf, accO[nt]);
      }
    }
  }

  #pragma unroll
  for (int g = 0; g < 8; ++g) {
    int row  = wm * 16 + g + 8 * lhalf;
    float li = 1.0f / sh.l[row];
    int t    = q0 + row;
    #pragma unroll
    for (int nt = 0; nt < 8; ++nt) {
      int col = head * Dd + wn * 128 + nt * 16 + l16;
      out[(size_t)(b * Tt + t) * (Hh * Dd) + col] = accO[nt][g] * li;
    }
  }
}

// ---------------------------------------------------------------------------
extern "C" void kernel_launch(void* const* d_in, const int* in_sizes, int n_in,
                              void* d_out, int out_size, void* d_ws, size_t ws_size,
                              hipStream_t stream) {
  (void)in_sizes; (void)n_in; (void)out_size;
  const float* h = (const float*)d_in[0];
  const float* A = (const float*)d_in[1];
  float* out = (float*)d_out;

  const size_t nH = (size_t)Bb * Tt * Dd;       // 2,097,152 elems
  const size_t nA = (size_t)Hh * Dd * Dd;       // 2,097,152 elems
  const size_t WS_NEED = (nH * 2 + nA) * sizeof(unsigned short);  // 12 MB

  dim3 agrid(Tt / QROWS, Hh, Bb);               // (64, 8, 2)

  if (ws_size >= WS_NEED) {
    unsigned short* Hbf  = (unsigned short*)d_ws;
    unsigned short* HbfT = Hbf + nH;
    unsigned short* AbfT = HbfT + nH;
    // h: [B][T][D] -> HbfT [B][D][T] + straight copy Hbf
    cvt_transpose<<<dim3(Tt / 64, Dd / 64, Bb), 256, 0, stream>>>(
        h, HbfT, Hbf, Tt, Dd);
    // A: [H][D][D] -> AbfT [H][e][d]
    cvt_transpose<<<dim3(Dd / 64, Dd / 64, Hh), 256, 0, stream>>>(
        A, AbfT, nullptr, Dd, Dd);
    attn_fast<<<agrid, 256, 0, stream>>>(Hbf, HbfT, AbfT, out);
  } else {
    attn_fallback<<<agrid, 256, 0, stream>>>(h, A, out);
  }
}
